// NaiveLSTM_3513283248844
// MI455X (gfx1250) — compile-verified, matrix operand order checked by reading, not run on hardware
//
#include <hip/hip_runtime.h>
#include <hip/hip_bf16.h>
#include <math.h>

typedef __bf16 bf16;
typedef __attribute__((ext_vector_type(16))) __bf16 v16bf;
typedef __attribute__((ext_vector_type(8)))  float  v8f;

#define BATCH 64
#define SEQ   512
#define INSZ  1024
#define HID   1024
#define G4    4096   // 4*HID
#define KDIM  1024
#define NBLOCKS_REC 32   // persistent recurrence blocks (must all be co-resident)

// ---------------- workspace layout (bytes) ----------------
#define OFF_XB    0            // B*S*I bf16          = 67,108,864
#define OFF_WHT   67108864     // 4096*1024 bf16      =  8,388,608
#define OFF_WXT   75497472     // 4096*1024 bf16      =  8,388,608
#define OFF_BIAS  83886080     // 4096 f32            =     16,384
#define OFF_GX    83902464     // B*S*4096 f32        = 536,870,912
#define OFF_C     620773376    // B*H f32             =    262,144
#define OFF_HMAX  621035520    // B*H f32             =    262,144
#define OFF_HINA  621297664    // B*H bf16            =    131,072
#define OFF_HINB  621428736    // B*H bf16            =    131,072
#define OFF_CTR   621559808    // barrier counter     =        256

__device__ __forceinline__ float sigmoid_f(float x) {
  return __builtin_amdgcn_rcpf(1.0f + __expf(-x));   // saturates cleanly: rcp(inf)=0
}
__device__ __forceinline__ float tanh_f(float x) {
  float e = __expf(2.0f * x);                        // limits exact at +-1
  return 1.0f - 2.0f * __builtin_amdgcn_rcpf(e + 1.0f);
}

// ---------------- fp32 -> bf16 convert of x ----------------
__global__ void __launch_bounds__(256) k_convert_x(const float* __restrict__ x,
                                                   bf16* __restrict__ xb, int n) {
  int i = blockIdx.x * 256 + threadIdx.x;
  if (i < n) xb[i] = (bf16)__builtin_nontemporal_load(&x[i]);
}

// ------- transpose+convert W -> WhT/WxT (N x K, K-contig), pack bias -------
// grid = (64 kTiles, 64 jTiles, 4 gates), block = 256
__global__ void __launch_bounds__(256) k_convert_w(
    const float* __restrict__ Wf, const float* __restrict__ Wi,
    const float* __restrict__ Wc, const float* __restrict__ Wo,
    const float* __restrict__ bf_, const float* __restrict__ bi_,
    const float* __restrict__ bc_, const float* __restrict__ bo_,
    bf16* __restrict__ WhT, bf16* __restrict__ WxT, float* __restrict__ bias) {
  __shared__ float sh[16][17];
  __shared__ float sx[16][17];
  int k0 = blockIdx.x * 16;
  int j0 = blockIdx.y * 16;
  int g  = blockIdx.z;
  const float* W  = (g == 0) ? Wf : (g == 1) ? Wi : (g == 2) ? Wc : Wo;
  const float* bg = (g == 0) ? bf_ : (g == 1) ? bi_ : (g == 2) ? bc_ : bo_;
  int tx = threadIdx.x & 15, ty = threadIdx.x >> 4;
  sh[ty][tx] = W[(k0 + ty) * HID + j0 + tx];          // h-rows (k < 1024)
  sx[ty][tx] = W[(INSZ + k0 + ty) * HID + j0 + tx];   // x-rows (k >= 1024)
  __syncthreads();
  int n = g * HID + j0 + ty;                          // output row (N index)
  WhT[n * KDIM + k0 + tx] = (bf16)sh[tx][ty];         // coalesced along k
  WxT[n * KDIM + k0 + tx] = (bf16)sx[tx][ty];
  if (blockIdx.x == 0 && ty == 0) bias[g * HID + j0 + tx] = bg[j0 + tx];
}

// ---------------- init recurrent state + barrier ----------------
__global__ void __launch_bounds__(256) k_init_state(float* __restrict__ Cst,
                                                    float* __restrict__ hmax,
                                                    bf16* __restrict__ hinA,
                                                    int* __restrict__ ctr) {
  int i = blockIdx.x * 256 + threadIdx.x;   // 65536 total
  Cst[i]  = 0.0f;
  hmax[i] = -INFINITY;      // running max starts at -inf
  hinA[i] = (bf16)0.0f;     // first-step hidden input is zeros
  if (i == 0) *ctr = 0;
}

// ------- Gx = x @ Wx + b : (32768 x 1024) @ (1024 x 4096) -------
// wave = 4 M-tiles x 4 gates; 32768 waves, 8 waves/block -> 4096 blocks
__global__ void __launch_bounds__(256) k_gx_gemm(const bf16* __restrict__ X,
                                                 const bf16* __restrict__ WxT,
                                                 const float* __restrict__ bias,
                                                 float* __restrict__ Gx) {
  int tid  = threadIdx.x;
  int lane = tid & 31;
  int w    = blockIdx.x * 8 + (tid >> 5);   // 0..32767
  int mQ   = w >> 6;                        // 0..511
  int jt   = w & 63;
  int mBase = mQ * 64;
  int jBase = jt * 16;
  int lm = lane & 15;
  int kh = (lane >> 4) * 16;                // K-half select: contiguous k = 16h + s, same for A and B

  const bf16* aP = X + (size_t)(mBase + lm) * KDIM + kh;
  const bf16* bP[4];
#pragma unroll
  for (int g = 0; g < 4; ++g)
    bP[g] = WxT + (size_t)(g * HID + jBase + lm) * KDIM + kh;

  v8f acc[4][4] = {};
  for (int kk = 0; kk < 32; ++kk) {
    int k0 = kk * 32;
    v16bf bfr[4];
#pragma unroll
    for (int g = 0; g < 4; ++g) bfr[g] = *(const v16bf*)(bP[g] + k0);
#pragma unroll
    for (int mt = 0; mt < 4; ++mt) {
      v16bf a = *(const v16bf*)(aP + mt * 16 * KDIM + k0);
#pragma unroll
      for (int g = 0; g < 4; ++g)
        acc[mt][g] = __builtin_amdgcn_wmma_f32_16x16x32_bf16(
            false, a, false, bfr[g], (short)0, acc[mt][g], false, false);
    }
  }
  int nCol   = jBase + lm;
  int rowOfs = 8 * (lane >> 4);
#pragma unroll
  for (int mt = 0; mt < 4; ++mt) {
#pragma unroll
    for (int r = 0; r < 8; ++r) {
      int row = mBase + mt * 16 + r + rowOfs;
#pragma unroll
      for (int g = 0; g < 4; ++g)
        __builtin_nontemporal_store(acc[mt][g][r] + bias[g * HID + nCol],
                                    &Gx[(size_t)row * G4 + g * HID + nCol]);
    }
  }
}

// ------- persistent recurrence: all 512 steps in one kernel -------
// wave = 1 M-tile x 4 gates; NBLOCKS_REC blocks x 256 threads = 256 waves
__global__ void __launch_bounds__(256) k_lstm_recurrence(
    bf16* __restrict__ hinA, bf16* __restrict__ hinB,
    const bf16* __restrict__ WhT, const float* __restrict__ Gx,
    float* __restrict__ Cst, float* __restrict__ hmax,
    float* __restrict__ out, int* barrier_ctr) {
  int tid  = threadIdx.x;
  int lane = tid & 31;
  int w    = blockIdx.x * 8 + (tid >> 5);   // 0..255
  int mt   = w >> 6;                        // 0..3
  int jt   = w & 63;
  int mBase = mt * 16;
  int jBase = jt * 16;
  int lm = lane & 15;
  int kh = (lane >> 4) * 16;

  const bf16* bP[4];
#pragma unroll
  for (int g = 0; g < 4; ++g)
    bP[g] = WhT + (size_t)(g * HID + jBase + lm) * KDIM + kh;
  int n      = jBase + lm;
  int rowOfs = 8 * (lane >> 4);

  for (int t = 0; t < SEQ; ++t) {
    const bf16* hr = (t & 1) ? hinB : hinA;
    bf16*       hw = (t & 1) ? hinA : hinB;
    const bf16* aP = hr + (mBase + lm) * KDIM + kh;

    v8f acc[4] = {};
    for (int kk = 0; kk < 32; ++kk) {
      int k0 = kk * 32;
      v16bf a = *(const v16bf*)(aP + k0);
#pragma unroll
      for (int g = 0; g < 4; ++g)
        acc[g] = __builtin_amdgcn_wmma_f32_16x16x32_bf16(
            false, a, false, *(const v16bf*)(bP[g] + k0), (short)0, acc[g], false, false);
    }

#pragma unroll
    for (int r = 0; r < 8; ++r) {
      int m = mBase + r + rowOfs;                       // batch index 0..63
      size_t gxB = (size_t)(m * SEQ + t) * G4;
      float gf = acc[0][r] + __builtin_nontemporal_load(&Gx[gxB + 0 * HID + n]);
      float gi = acc[1][r] + __builtin_nontemporal_load(&Gx[gxB + 1 * HID + n]);
      float gc = acc[2][r] + __builtin_nontemporal_load(&Gx[gxB + 2 * HID + n]);
      float go = acc[3][r] + __builtin_nontemporal_load(&Gx[gxB + 3 * HID + n]);
      float f = sigmoid_f(gf);
      float i = sigmoid_f(gi);
      float c = tanh_f(gc);
      float o = sigmoid_f(go);
      int sidx = m * HID + n;
      float Cn = f * Cst[sidx] + i * c;
      Cst[sidx] = Cn;
      float h = o * tanh_f(Cn);
      __builtin_nontemporal_store(h, &out[(size_t)(m * SEQ + t) * HID + n]);
      float hm = fmaxf(hmax[sidx], h);
      hmax[sidx]  = hm;                                 // running cummax
      hw[sidx]    = (bf16)hm;                           // next-step input hidden
    }

    // ---- device-wide barrier between steps ----
    __threadfence();                                    // flush our stores (agent scope)
    __syncthreads();
    if (tid == 0) {
      __hip_atomic_fetch_add(barrier_ctr, 1, __ATOMIC_ACQ_REL, __HIP_MEMORY_SCOPE_AGENT);
      int target = NBLOCKS_REC * (t + 1);
      while (__hip_atomic_load(barrier_ctr, __ATOMIC_ACQUIRE, __HIP_MEMORY_SCOPE_AGENT) < target)
        __builtin_amdgcn_s_sleep(1);
    }
    __syncthreads();
    __threadfence();                                    // invalidate stale L0 before reading hin
  }
}

// ---------------- tail: h_fin (=hmax), C_fin ----------------
__global__ void __launch_bounds__(256) k_finalize(const float* __restrict__ hmax,
                                                  const float* __restrict__ Cst,
                                                  float* __restrict__ out) {
  int i = blockIdx.x * 256 + threadIdx.x;   // 65536
  const size_t base = (size_t)BATCH * SEQ * HID;
  out[base + i]               = hmax[i];
  out[base + BATCH * HID + i] = Cst[i];
}

extern "C" void kernel_launch(void* const* d_in, const int* in_sizes, int n_in,
                              void* d_out, int out_size, void* d_ws, size_t ws_size,
                              hipStream_t stream) {
  const float* x  = (const float*)d_in[0];
  const float* Wf = (const float*)d_in[1]; const float* bf_ = (const float*)d_in[2];
  const float* Wi = (const float*)d_in[3]; const float* bi_ = (const float*)d_in[4];
  const float* Wc = (const float*)d_in[5]; const float* bc_ = (const float*)d_in[6];
  const float* Wo = (const float*)d_in[7]; const float* bo_ = (const float*)d_in[8];

  char* ws = (char*)d_ws;
  bf16*  xb   = (bf16*)(ws + OFF_XB);
  bf16*  WhT  = (bf16*)(ws + OFF_WHT);
  bf16*  WxT  = (bf16*)(ws + OFF_WXT);
  float* bias = (float*)(ws + OFF_BIAS);
  float* Gx   = (float*)(ws + OFF_GX);
  float* Cst  = (float*)(ws + OFF_C);
  float* hmax = (float*)(ws + OFF_HMAX);
  bf16*  hinA = (bf16*)(ws + OFF_HINA);
  bf16*  hinB = (bf16*)(ws + OFF_HINB);
  int*   ctr  = (int*)(ws + OFF_CTR);
  float* out  = (float*)d_out;

  const int nX = BATCH * SEQ * INSZ;  // 33,554,432
  k_convert_x<<<(nX + 255) / 256, 256, 0, stream>>>(x, xb, nX);
  k_convert_w<<<dim3(64, 64, 4), 256, 0, stream>>>(Wf, Wi, Wc, Wo, bf_, bi_, bc_, bo_,
                                                   WhT, WxT, bias);
  k_init_state<<<256, 256, 0, stream>>>(Cst, hmax, hinA, ctr);
  k_gx_gemm<<<4096, 256, 0, stream>>>(xb, WxT, bias, Gx);
  k_lstm_recurrence<<<NBLOCKS_REC, 256, 0, stream>>>(hinA, hinB, WhT, Gx, Cst, hmax,
                                                     out, ctr);
  k_finalize<<<256, 256, 0, stream>>>(hmax, Cst, out);
}
